// Quanvolution__gen148_65481071407565
// MI455X (gfx1250) — matrix operand order repaired and checked by reading, not compiled, not run on hardware
//
#include <hip/hip_runtime.h>
#include <hip/hip_bf16.h>
#include <stddef.h>

typedef __attribute__((ext_vector_type(16))) _Float16 v16h;
typedef __attribute__((ext_vector_type(8)))  _Float16 v8h;
typedef __attribute__((ext_vector_type(4)))  _Float16 v4h;
typedef __attribute__((ext_vector_type(8)))  float    v8f;

#define NQ    4
#define NB    14
#define NP    196      // NB*NB patches per image
#define IMG   28
#define BATCH 2048
#define KDIM  784      // NP*NQ features
#define KT    25       // K tiles of 32 -> Kpad = 800
#define NCLS  10

// ---------------- 4-qubit gate helpers (compile-time masks => all-register state) ----

template<int MASK>
__device__ __forceinline__ void apply_ry(float c, float s, float* sr, float* si) {
#pragma unroll
  for (int i = 0; i < 16; ++i) {
    if (!(i & MASK)) {
      const int j = i | MASK;
      float r0 = sr[i], r1 = sr[j], q0 = si[i], q1 = si[j];
      sr[i] = c * r0 - s * r1;  sr[j] = s * r0 + c * r1;
      si[i] = c * q0 - s * q1;  si[j] = s * q0 + c * q1;
    }
  }
}

template<int MASK>
__device__ __forceinline__ void apply_rz(float c, float s, float* sr, float* si) {
#pragma unroll
  for (int i = 0; i < 16; ++i) {
    float r = sr[i], q = si[i];
    if (i & MASK) { sr[i] = c * r - s * q;  si[i] = c * q + s * r; }   // * conj(ph)
    else          { sr[i] = c * r + s * q;  si[i] = c * q - s * r; }   // * ph = e^{-i t/2}
  }
}

template<int CM, int TM>
__device__ __forceinline__ void apply_cnot(float* sr, float* si) {
#pragma unroll
  for (int i = 0; i < 16; ++i) {
    if ((i & CM) && !(i & TM)) {
      const int j = i | TM;
      float t = sr[i]; sr[i] = sr[j]; sr[j] = t;
      t = si[i]; si[i] = si[j]; si[j] = t;
    }
  }
}

// ---------------- kernel 0: swizzle W (10x784 f32) into WMMA B fragments (f16) -------
// B layout (16-bit, 32x16 K x N): lane = n + 16*((k_in_tile>>3)&1), half = (k_in_tile&7) + 8*(k_in_tile>=16).
__global__ __launch_bounds__(256) void quanv_wprep_kernel(
    const float* __restrict__ W, _Float16* __restrict__ wfrag) {
  const int t = blockIdx.x * 256 + threadIdx.x;     // 0 .. 25*32*16-1 = 12799
  if (t >= KT * 32 * 16) return;
  const int hidx = t & 15;
  const int lane = (t >> 4) & 31;
  const int kt   = t >> 9;
  const int n = lane & 15;
  const int k = kt * 32 + ((hidx >> 3) << 4) + ((lane >> 4) << 3) + (hidx & 7);
  const float v = (n < NCLS && k < KDIM) ? W[n * KDIM + k] : 0.f;
  wfrag[t] = (_Float16)v;
}

// ---------------- kernel 1: simulate one 4-qubit circuit per thread -----------------
// Wire w owns state-index bit (8>>w). Writes 4 features straight into the WMMA
// A-fragment layout (16-bit A 16x32): lane = m + 16*((k>>3)&1), half = (k&7)+8*(k>=16).
__global__ __launch_bounds__(256) void quanv_sim_kernel(
    const float* __restrict__ x, const float* __restrict__ vp,
    _Float16* __restrict__ afrag) {
  const int idx = blockIdx.x * 256 + threadIdx.x;
  if (idx >= BATCH * NP) return;
  const int b  = idx / NP;
  const int p  = idx - b * NP;
  const int pr = p / NB;
  const int pc = p - pr * NB;

  const float* xb = x + (size_t)b * (IMG * IMG) + (2 * pr) * IMG + 2 * pc;
  const float a0 = xb[0], a1 = xb[1], a2 = xb[IMG], a3 = xb[IMG + 1];

  float sr[16], si[16];
#pragma unroll
  for (int i = 0; i < 16; ++i) { sr[i] = 0.f; si[i] = 0.f; }
  sr[0] = 1.f;

  float s, c;
  // encoding RY per wire
  __sincosf(0.5f * a0, &s, &c); apply_ry<8>(c, s, sr, si);
  __sincosf(0.5f * a1, &s, &c); apply_ry<4>(c, s, sr, si);
  __sincosf(0.5f * a2, &s, &c); apply_ry<2>(c, s, sr, si);
  __sincosf(0.5f * a3, &s, &c); apply_ry<1>(c, s, sr, si);

#pragma unroll
  for (int l = 0; l < 2; ++l) {
    const float* v = vp + l * 8;            // (NL,2,NQ) row-major
    __sincosf(0.5f * v[0], &s, &c); apply_ry<8>(c, s, sr, si);
    __sincosf(0.5f * v[1], &s, &c); apply_ry<4>(c, s, sr, si);
    __sincosf(0.5f * v[2], &s, &c); apply_ry<2>(c, s, sr, si);
    __sincosf(0.5f * v[3], &s, &c); apply_ry<1>(c, s, sr, si);
    __sincosf(0.5f * v[4], &s, &c); apply_rz<8>(c, s, sr, si);
    __sincosf(0.5f * v[5], &s, &c); apply_rz<4>(c, s, sr, si);
    __sincosf(0.5f * v[6], &s, &c); apply_rz<2>(c, s, sr, si);
    __sincosf(0.5f * v[7], &s, &c); apply_rz<1>(c, s, sr, si);
    apply_cnot<8, 4>(sr, si);
    apply_cnot<4, 2>(sr, si);
    apply_cnot<2, 1>(sr, si);
  }

  float e0 = 0.f, e1 = 0.f, e2 = 0.f, e3 = 0.f;
#pragma unroll
  for (int i = 0; i < 16; ++i) {
    const float pw = sr[i] * sr[i] + si[i] * si[i];
    e0 += (i & 8) ? -pw : pw;
    e1 += (i & 4) ? -pw : pw;
    e2 += (i & 2) ? -pw : pw;
    e3 += (i & 1) ? -pw : pw;
  }

  // scatter into A-fragment layout
  const int btile = b >> 4, mlane = b & 15;
  const int k0 = p * 4;
  const int kt = k0 >> 5;
  const int r  = k0 & 31;
  const int lane = mlane + (((r >> 3) & 1) << 4);
  const int hidx = (r & 7) + ((r & 16) ? 8 : 0);
  _Float16* dst = afrag + (((size_t)btile * KT + kt) * 32 + lane) * 16 + hidx;
  v4h o; o.x = (_Float16)e0; o.y = (_Float16)e1; o.z = (_Float16)e2; o.w = (_Float16)e3;
  *(v4h*)dst = o;

  // zero the K padding (k = 784..799 lives in tile 24, halves 8..15 of both lane halves)
  if (p < 2) {
    const int plane = mlane + (p << 4);
    _Float16* pd = afrag + (((size_t)btile * KT + 24) * 32 + plane) * 16 + 8;
    v8h z = {};
    *(v8h*)pd = z;
  }
}

// ---------------- kernel 2: WMMA GEMM (16x16 tile, K=800) + bias + log-softmax ------
__global__ __launch_bounds__(32) void quanv_gemm_kernel(
    const _Float16* __restrict__ afrag, const _Float16* __restrict__ wfrag,
    const float* __restrict__ bias, float* __restrict__ out) {
  const int lane  = threadIdx.x;
  const int btile = blockIdx.x;

  v8f acc = {};
  const v16h* ap = (const v16h*)afrag + (size_t)btile * KT * 32 + lane;
  const v16h* bp = (const v16h*)wfrag + lane;
#pragma unroll
  for (int kt = 0; kt < KT; ++kt) {
    const v16h a  = ap[kt * 32];
    const v16h bm = bp[kt * 32];
    acc = __builtin_amdgcn_wmma_f32_16x16x32_f16(
        /*neg_a=*/false, a, /*neg_b=*/false, bm,
        /*c_mod=*/(short)0, acc, /*reuse_a=*/false, /*reuse_b=*/false);
  }

  // D layout: VGPR v -> row (lane<16 ? v : v+8), col = lane&15
  __shared__ float tile[16][16];
#pragma unroll
  for (int v = 0; v < 8; ++v) {
    const int m = (lane < 16) ? v : (v + 8);
    tile[m][lane & 15] = acc[v];
  }
  __syncthreads();

  if (lane < 16) {
    const int row = lane;
    float lg[NCLS];
    float mx = -1e30f;
#pragma unroll
    for (int n = 0; n < NCLS; ++n) { lg[n] = tile[row][n] + bias[n]; mx = fmaxf(mx, lg[n]); }
    float se = 0.f;
#pragma unroll
    for (int n = 0; n < NCLS; ++n) se += __expf(lg[n] - mx);
    const float ls = __logf(se);
    float* o = out + ((size_t)btile * 16 + row) * NCLS;
#pragma unroll
    for (int n = 0; n < NCLS; ++n) o[n] = lg[n] - mx - ls;
  }
}

// ---------------- launcher ----------------------------------------------------------
extern "C" void kernel_launch(void* const* d_in, const int* in_sizes, int n_in,
                              void* d_out, int out_size, void* d_ws, size_t ws_size,
                              hipStream_t stream) {
  const float* x    = (const float*)d_in[0];   // 2048*28*28
  const float* vp   = (const float*)d_in[1];   // 2*2*4
  const float* W    = (const float*)d_in[2];   // 10*784
  const float* bias = (const float*)d_in[3];   // 10
  float* out = (float*)d_out;                  // 2048*10

  _Float16* wfrag = (_Float16*)d_ws;                       // 25*32*16 halves = 25,600 B
  _Float16* afrag = (_Float16*)((char*)d_ws + 32768);      // 128*25*32*16 halves ≈ 3.1 MB

  quanv_wprep_kernel<<<(KT * 32 * 16 + 255) / 256, 256, 0, stream>>>(W, wfrag);
  quanv_sim_kernel<<<(BATCH * NP) / 256, 256, 0, stream>>>(x, vp, afrag);
  quanv_gemm_kernel<<<BATCH / 16, 32, 0, stream>>>(afrag, wfrag, bias, out);
}
